// DiscourseTransformer_57543971832184
// MI455X (gfx1250) — matrix operand order repaired and checked by reading, not compiled, not run on hardware
//
#include <hip/hip_runtime.h>
#include <hip/hip_bf16.h>
#include <math.h>

// ---------------- model constants ----------------
#define B_   4
#define T_   12
#define LL   64
#define HH   256
#define NHH  8
#define HD_  32
#define FF   1024
#define SM_  8

// ---------------- WMMA types ----------------
typedef __attribute__((ext_vector_type(16))) __bf16 v16bf;
typedef __attribute__((ext_vector_type(8)))  float  v8f;

union BF16x16 { v16bf v; unsigned u[8]; };

__device__ __forceinline__ unsigned pack_bf16(float a, float b) {
  unsigned ua = __float_as_uint(a), ub = __float_as_uint(b);
  ua = ua + 0x7FFFu + ((ua >> 16) & 1u);         // RNE to bf16
  ub = ub + 0x7FFFu + ((ub >> 16) & 1u);
  return (ua >> 16) | (ub & 0xFFFF0000u);
}

// A-matrix (16x32 bf16) K offset for VGPR v, lane-half hi (ISA 7.12.2)
__device__ __forceinline__ int kb_a(int v, int hi) {
  return (v < 4 ? 2 * v : 16 + 2 * (v - 4)) + (hi ? 8 : 0);
}
// B-matrix (32x16 bf16) K offset: same K order as kb_a
__device__ __forceinline__ int kb_b(int v, int hi) {
  return (v < 4 ? 2 * v : 16 + 2 * (v - 4)) + (hi ? 8 : 0);
}

__device__ __forceinline__ float wave_sum(float v) {
#pragma unroll
  for (int o = 16; o; o >>= 1) v += __shfl_xor(v, o, 32);
  return v;
}
__device__ __forceinline__ float wave_max(float v) {
#pragma unroll
  for (int o = 16; o; o >>= 1) v = fmaxf(v, __shfl_xor(v, o, 32));
  return v;
}
// reductions within each 16-lane half (matches WMMA D-layout row split)
__device__ __forceinline__ float half_sum(float v) {
#pragma unroll
  for (int o = 8; o; o >>= 1) v += __shfl_xor(v, o, 32);
  return v;
}
__device__ __forceinline__ float half_max(float v) {
#pragma unroll
  for (int o = 8; o; o >>= 1) v = fmaxf(v, __shfl_xor(v, o, 32));
  return v;
}

// ---------------- utility kernels ----------------
__global__ void k_zerof(float* p, int n) {
  for (int i = blockIdx.x * blockDim.x + threadIdx.x; i < n; i += gridDim.x * blockDim.x)
    p[i] = 0.f;
}

__global__ void k_cvt_bf16(const float* __restrict__ src, unsigned short* __restrict__ dst, int n) {
  for (int i = blockIdx.x * blockDim.x + threadIdx.x; i < n; i += gridDim.x * blockDim.x) {
    unsigned u = __float_as_uint(src[i]);
    u = u + 0x7FFFu + ((u >> 16) & 1u);
    dst[i] = (unsigned short)(u >> 16);
  }
}

// sel[b,t,p,h] = tok_emb[ids[b,t,p]][h] + tok_pe[p][h] + spk_emb[spk[b,t]][h]
__global__ void k_embed(const int* __restrict__ ids, const int* __restrict__ spk,
                        const float* __restrict__ tok_emb, const float* __restrict__ tok_pe,
                        const float* __restrict__ spk_emb, float* __restrict__ sel) {
  size_t total = (size_t)B_ * T_ * LL * HH;
  for (size_t i = (size_t)blockIdx.x * blockDim.x + threadIdx.x; i < total;
       i += (size_t)gridDim.x * blockDim.x) {
    int h = (int)(i & (HH - 1));
    size_t r = i >> 8;
    int p = (int)(r & (LL - 1));
    size_t bt = r >> 6;
    int tok = ids[bt * LL + p];
    int s   = spk[bt];
    sel[i] = tok_emb[(size_t)tok * HH + h] + tok_pe[(size_t)p * HH + h] + spk_emb[(size_t)s * HH + h];
  }
}

__global__ void k_slice_cur(const float* __restrict__ sel, int l, float* __restrict__ out) {
  size_t total = (size_t)B_ * LL * HH;
  for (size_t i = (size_t)blockIdx.x * blockDim.x + threadIdx.x; i < total;
       i += (size_t)gridDim.x * blockDim.x) {
    int h = (int)(i & (HH - 1));
    size_t r = i >> 8;
    int p = (int)(r & (LL - 1));
    int b = (int)(r >> 6);
    out[i] = sel[((((size_t)b * T_) + l) * LL + p) * HH + h];
  }
}

// rel + edu_pe, reshaped to (B*lp, 128, H)
__global__ void k_build_rel(const float* __restrict__ sel, const float* __restrict__ edu_pe,
                            float* __restrict__ y, int l, int lp) {
  size_t total = (size_t)B_ * lp * 128 * HH;
  for (size_t i = (size_t)blockIdx.x * blockDim.x + threadIdx.x; i < total;
       i += (size_t)gridDim.x * blockDim.x) {
    int h = (int)(i & (HH - 1));
    size_t rt = i >> 8;
    int t = (int)(rt & 127);
    size_t bj = rt >> 7;
    int j = (int)(bj % lp);
    int b = (int)(bj / lp);
    float base; int dist;
    if (t < LL) {
      int src_l = (l == 0) ? 0 : j;
      base = sel[((((size_t)b * T_) + src_l) * LL + t) * HH + h];
      dist = (l == 0) ? 0 : (l - j);               // <= 11 < EMAX-1
    } else {
      base = sel[((((size_t)b * T_) + l) * LL + (t - LL)) * HH + h];
      dist = 0;
    }
    y[i] = base + edu_pe[(size_t)dist * HH + h];
  }
}

// ---------------- WMMA GEMM: D[M,N] = act(A[M,K] @ Wbf16[N,K]^T + bias) ----------------
// block = 128 threads (4 waves); wave w covers n0 = blockIdx.y*256 + w*64 .. +63
// A tile (16xK) converted to bf16 ONCE per block into LDS; per wave, fragments are
// assembled with ds_load's. K loop is software-pipelined 2 deep (ping-pong a0/b0,
// a1/b1) so loads for step k+64/k+96 overlap the WMMAs of k/k+32. K in {256,1024}.
__global__ void k_gemm_bias(const float* __restrict__ A, const unsigned short* __restrict__ Wb,
                            const float* __restrict__ bias, float* __restrict__ D,
                            int M, int N, int K, int act) {
  int lane = threadIdx.x & 31;
  int wave = threadIdx.x >> 5;
  int hi = lane >> 4;
  int lm = lane & 15;
  int m0 = blockIdx.x * 16;
  int n0 = blockIdx.y * 256 + wave * 64;
  int hi4 = hi * 4, hi8 = hi * 8;

  // stage A tile (16 rows x K) as packed bf16 pairs in LDS
  __shared__ __align__(16) unsigned Asm[16 * (FF / 2)];   // 32 KB max (K<=1024)
  int kd2 = K >> 1;  // dwords per row
  for (int r = 0; r < 16; ++r) {
    int m = m0 + r; if (m >= M) m = M - 1;                // clamp: garbage rows never stored
    const float* Ar = A + (size_t)m * K;
    unsigned* dst = Asm + r * kd2;
    for (int d = threadIdx.x; d < kd2; d += 128) {
      float2 p = *(const float2*)(Ar + 2 * d);
      dst[d] = pack_bf16(p.x, p.y);
    }
  }
  __syncthreads();

  __builtin_prefetch(Wb + (size_t)n0 * K, 0, 1);          // global_prefetch_b8

  const unsigned* Amy = Asm + lm * kd2;
  const unsigned* Wrow[4];
#pragma unroll
  for (int nt = 0; nt < 4; ++nt)
    Wrow[nt] = (const unsigned*)(Wb + (size_t)(n0 + nt * 16 + lm) * K);

  v8f acc[4] = {};
  BF16x16 a0, a1, b0[4], b1[4];

  auto load_step = [&](int kk, BF16x16& aa, BF16x16 (&bb)[4]) {
    int kd = kk >> 1;
#pragma unroll
    for (int v = 0; v < 8; ++v)
      aa.u[v] = Amy[kd + v + hi8];                        // 8 consecutive dwords
#pragma unroll
    for (int nt = 0; nt < 4; ++nt)
#pragma unroll
      for (int v = 0; v < 8; ++v)
        bb[nt].u[v] = Wrow[nt][kd + v + hi8];             // 8 consecutive dwords
  };
  auto mma4 = [&](BF16x16& aa, BF16x16 (&bb)[4]) {
#pragma unroll
    for (int nt = 0; nt < 4; ++nt)
      acc[nt] = __builtin_amdgcn_wmma_f32_16x16x32_bf16(false, aa.v, false, bb[nt].v,
                                                        (short)0, acc[nt], false, false);
  };

  load_step(0, a0, b0);
  load_step(32, a1, b1);
  int kk = 0;
  for (; kk + 64 < K; kk += 64) {
    mma4(a0, b0); load_step(kk + 64, a0, b0);
    mma4(a1, b1); load_step(kk + 96, a1, b1);
  }
  mma4(a0, b0);
  mma4(a1, b1);

#pragma unroll
  for (int nt = 0; nt < 4; ++nt) {
    int n = n0 + nt * 16 + lm;
    float bv = bias ? bias[n] : 0.f;
#pragma unroll
    for (int r = 0; r < 8; ++r) {
      int m = m0 + r + hi * 8;
      if (m < M) {
        float x = acc[nt][r] + bv;
        if (act == 1)      x = 0.5f * x * (1.f + erff(x * 0.70710678118654752f)); // exact gelu
        else if (act == 2) x = tanhf(x);
        D[(size_t)m * N + n] = x;
      }
    }
  }
}

// ---------------- WMMA attention (one wave: 16 query rows x one head) ----------------
// qkv row layout: [Q(256) | K(256) | V(256)], head h at offset h*32 within each.
template <int S>
__global__ void k_attn(const float* __restrict__ qkv, float* __restrict__ outp) {
  constexpr int NT = S / 16;
  int lane = threadIdx.x & 31;
  int hi = lane >> 4;
  int lm = lane & 15;
  int head = blockIdx.x & (NHH - 1);
  int seq  = blockIdx.x >> 3;
  int q0 = blockIdx.y * 16;
  __shared__ __align__(16) float sm[16 * S];
  const float* base = qkv + (size_t)seq * S * (3 * HH);

  BF16x16 aq;
  {
    const float* qrow = base + (size_t)(q0 + lm) * (3 * HH) + head * HD_;
#pragma unroll
    for (int v = 0; v < 8; ++v) { int kb = kb_a(v, hi); aq.u[v] = pack_bf16(qrow[kb], qrow[kb + 1]); }
  }
  v8f acc[NT];
#pragma unroll
  for (int nt = 0; nt < NT; ++nt) {
    BF16x16 bk;
    const float* krow = base + (size_t)(nt * 16 + lm) * (3 * HH) + HH + head * HD_;
#pragma unroll
    for (int v = 0; v < 8; ++v) { int kb = kb_b(v, hi); bk.u[v] = pack_bf16(krow[kb], krow[kb + 1]); }
    v8f z = {};
    acc[nt] = __builtin_amdgcn_wmma_f32_16x16x32_bf16(false, aq.v, false, bk.v,
                                                      (short)0, z, false, false);
  }
  const float scale = 0.17677669529663688f;  // 1/sqrt(32)
#pragma unroll
  for (int r = 0; r < 8; ++r) {
    float mx = -1e30f;
#pragma unroll
    for (int nt = 0; nt < NT; ++nt) { acc[nt][r] *= scale; mx = fmaxf(mx, acc[nt][r]); }
    mx = half_max(mx);
    float s = 0.f;
#pragma unroll
    for (int nt = 0; nt < NT; ++nt) { float e = __expf(acc[nt][r] - mx); acc[nt][r] = e; s += e; }
    s = half_sum(s);
    float inv = 1.f / s;
#pragma unroll
    for (int nt = 0; nt < NT; ++nt) {
      float p = acc[nt][r] * inv;
      sm[(r + hi * 8) * S + nt * 16 + lm] = p;   // LDS bounce: D-layout -> row-major
    }
  }
  __syncthreads();

  v8f o0 = {}, o1 = {};
  for (int ks = 0; ks < S; ks += 32) {
    BF16x16 ap;
#pragma unroll
    for (int v = 0; v < 8; ++v) {
      int kb = kb_a(v, hi);
      ap.u[v] = pack_bf16(sm[lm * S + ks + kb], sm[lm * S + ks + kb + 1]);
    }
#pragma unroll
    for (int nd = 0; nd < 2; ++nd) {
      BF16x16 bv;
      int d = nd * 16 + lm;
#pragma unroll
      for (int v = 0; v < 8; ++v) {
        int kb = kb_b(v, hi);
        const float* vp = base + (size_t)(ks + kb) * (3 * HH) + 2 * HH + head * HD_ + d;
        bv.u[v] = pack_bf16(vp[0], vp[3 * HH]);
      }
      if (nd == 0)
        o0 = __builtin_amdgcn_wmma_f32_16x16x32_bf16(false, ap.v, false, bv.v, (short)0, o0, false, false);
      else
        o1 = __builtin_amdgcn_wmma_f32_16x16x32_bf16(false, ap.v, false, bv.v, (short)0, o1, false, false);
    }
  }
#pragma unroll
  for (int r = 0; r < 8; ++r) {
    int m = q0 + r + hi * 8;
    float* orow = outp + ((size_t)seq * S + m) * HH + head * HD_;
    orow[lm]      = o0[r];
    orow[16 + lm] = o1[r];
  }
}

// ---------------- LayerNorm: out = LN(a + c) * g + be, one row per block (256 thr) ----------------
__global__ void k_add_ln(const float* __restrict__ a, const float* __restrict__ c,
                         const float* __restrict__ g, const float* __restrict__ be,
                         float* __restrict__ out) {
  int m = blockIdx.x, t = threadIdx.x;
  size_t idx = (size_t)m * HH + t;
  float v = a[idx] + c[idx];
  __shared__ float red[8];
  __shared__ float sval[2];
  float s = wave_sum(v);
  if ((t & 31) == 0) red[t >> 5] = s;
  __syncthreads();
  if (t == 0) { float x = 0; for (int i = 0; i < 8; ++i) x += red[i]; sval[0] = x * (1.f / HH); }
  __syncthreads();
  float mean = sval[0];
  float d = v - mean;
  float q = wave_sum(d * d);
  if ((t & 31) == 0) red[t >> 5] = q;
  __syncthreads();
  if (t == 0) { float x = 0; for (int i = 0; i < 8; ++i) x += red[i]; sval[1] = x * (1.f / HH); }
  __syncthreads();
  out[idx] = g[t] * d * rsqrtf(sval[1] + 1e-5f) + be[t];
}

// ---------------- speaker scoring ----------------
__global__ void k_spk_scores(const float* __restrict__ mem, const float* __restrict__ logits,
                             float* __restrict__ scores) {
  int b = blockIdx.x / SM_, s = blockIdx.x % SM_;
  int t = threadIdx.x;  // 0..63 over tokens
  const float* mv = mem + ((size_t)b * SM_ + s) * HH;
  const float* lg = logits + ((size_t)b * LL + t) * HH;
  float d = 0.f;
  for (int h = 0; h < HH; ++h) d += mv[h] * lg[h];
  float mx = wave_max(d);
  __shared__ float red[2];
  if ((t & 31) == 0) red[t >> 5] = mx;
  __syncthreads();
  if (t == 0) scores[b * SM_ + s] = fmaxf(red[0], red[1]);
}

__global__ void k_spk_soft(const float* __restrict__ scores, const int* __restrict__ spk, int l,
                           float* __restrict__ pspk, float* __restrict__ flop) {
  int b = threadIdx.x;
  if (b < B_) {
    const float* sc = scores + b * SM_;
    float mx = -1e30f;
    for (int s = 0; s < SM_; ++s) mx = fmaxf(mx, sc[s]);
    float e[SM_]; float sum = 0.f;
    for (int s = 0; s < SM_; ++s) { e[s] = __expf(sc[s] - mx); sum += e[s]; }
    int sid = spk[b * T_ + l];
    float ent = 0.f, ps = 0.f;
    for (int s = 0; s < SM_; ++s) {
      float p = e[s] / sum;
      ent -= p * logf(p + 1e-12f);
      if (s == sid) ps = p;
    }
    pspk[b] = ps;
    atomicAdd(flop, ent * (1.f / B_));
  }
}

// ---------------- mean over lp, write to output slice + ws ----------------
__global__ void k_ymean(const float* __restrict__ y, int lp, int l,
                        float* __restrict__ dout, float* __restrict__ ymean) {
  size_t total = (size_t)B_ * LL * HH;
  float invlp = 1.f / (float)lp;
  for (size_t i = (size_t)blockIdx.x * blockDim.x + threadIdx.x; i < total;
       i += (size_t)gridDim.x * blockDim.x) {
    int h = (int)(i & (HH - 1));
    size_t r = i >> 8;
    int t = (int)(r & (LL - 1));
    int b = (int)(r >> 6);
    float s = 0.f;
    for (int j = 0; j < lp; ++j)
      s += y[(((size_t)(b * lp + j)) * 128 + LL + t) * HH + h];
    s *= invlp;
    ymean[i] = s;
    dout[((((size_t)b * T_) + l) * LL + t) * HH + h] = s;
  }
}

// ---------------- pooling ----------------
__global__ void k_pool_score(const float* __restrict__ tanhbuf, const float* __restrict__ pv,
                             const int* __restrict__ amask, int l, float* __restrict__ alpha) {
  int b = blockIdx.x, t = threadIdx.x;  // 64 threads
  const float* row = tanhbuf + ((size_t)b * LL + t) * HH;
  float d = 0.f;
  for (int h = 0; h < HH; ++h) d += row[h] * pv[h];
  if (!amask[(((size_t)b * T_) + l) * LL + t]) d = -10000.f;
  __shared__ float shm[2], shs[2];
  float mx = wave_max(d);
  if ((t & 31) == 0) shm[t >> 5] = mx;
  __syncthreads();
  float bmx = fmaxf(shm[0], shm[1]);
  float e = __expf(d - bmx);
  float s = wave_sum(e);
  if ((t & 31) == 0) shs[t >> 5] = s;
  __syncthreads();
  alpha[b * LL + t] = e / (shs[0] + shs[1]);
}

__global__ void k_pool_combine(const float* __restrict__ alpha, const float* __restrict__ ymean,
                               float* __restrict__ edu) {
  int b = blockIdx.x, h = threadIdx.x;  // 256 threads
  float s = 0.f;
  for (int t = 0; t < LL; ++t)
    s += alpha[b * LL + t] * ymean[((size_t)b * LL + t) * HH + h];
  edu[b * HH + h] = s;
}

__global__ void k_updprob(const float* __restrict__ edu, const float* __restrict__ uW,
                          const float* __restrict__ ub, float* __restrict__ pcnt) {
  int b = blockIdx.x, t = threadIdx.x;
  float v = edu[b * HH + t] * uW[t];
  __shared__ float red[8];
  float s = wave_sum(v);
  if ((t & 31) == 0) red[t >> 5] = s;
  __syncthreads();
  if (t == 0) {
    float x = 0; for (int i = 0; i < 8; ++i) x += red[i];
    pcnt[b] = 1.f / (1.f + __expf(-(x + ub[0])));
  }
}

// ---------------- GRU + memory update ----------------
__global__ void k_gather_hold(const float* __restrict__ mem, const int* __restrict__ spk, int l,
                              float* __restrict__ hold) {
  int b = blockIdx.x, h = threadIdx.x;
  int sid = spk[b * T_ + l];
  hold[b * HH + h] = mem[((size_t)b * SM_ + sid) * HH + h];
}

__global__ void k_gru(const float* __restrict__ gi, const float* __restrict__ gh,
                      const float* __restrict__ hold, const float* __restrict__ pspk,
                      const float* __restrict__ pcnt, const int* __restrict__ spk, int l,
                      float* __restrict__ mem) {
  int b = blockIdx.x, h = threadIdx.x;
  float ir = gi[b * 768 + h], iz = gi[b * 768 + 256 + h], inn = gi[b * 768 + 512 + h];
  float hr = gh[b * 768 + h], hz = gh[b * 768 + 256 + h], hn = gh[b * 768 + 512 + h];
  float r = 1.f / (1.f + __expf(-(ir + hr)));
  float z = 1.f / (1.f + __expf(-(iz + hz)));
  float n = tanhf(inn + r * hn);
  float ho = hold[b * HH + h];
  float hnew = (1.f - z) * n + z * ho;
  bool upd = (pspk[b] * pcnt[b]) >= 0.5f;
  int sid = spk[b * T_ + l];
  mem[((size_t)b * SM_ + sid) * HH + h] = upd ? hnew : ho;
}

__global__ void k_finalize(const float* __restrict__ flop, float* __restrict__ out) {
  if (threadIdx.x == 0 && blockIdx.x == 0) out[0] = flop[0] * (1.f / T_);
}

// ---------------- host orchestration ----------------
extern "C" void kernel_launch(void* const* d_in, const int* in_sizes, int n_in,
                              void* d_out, int out_size, void* d_ws, size_t ws_size,
                              hipStream_t stream) {
  (void)in_sizes; (void)n_in; (void)out_size; (void)ws_size;
  const int* ids     = (const int*)d_in[0];
  const int* amask   = (const int*)d_in[1];
  const int* spk     = (const int*)d_in[2];
  const float* tok_emb = (const float*)d_in[3];
  const float* tok_pe  = (const float*)d_in[4];
  const float* edu_pe  = (const float*)d_in[5];
  const float* spk_emb = (const float*)d_in[6];

  struct LP { const float *Wqkv, *bqkv, *Wo, *bo, *W1, *bias1, *W2, *bias2, *g1, *b1, *g2, *b2; };
  auto getLP = [&](int base) {
    LP w;
    w.Wqkv = (const float*)d_in[base + 0];  w.bqkv  = (const float*)d_in[base + 1];
    w.Wo   = (const float*)d_in[base + 2];  w.bo    = (const float*)d_in[base + 3];
    w.W1   = (const float*)d_in[base + 4];  w.bias1 = (const float*)d_in[base + 5];
    w.W2   = (const float*)d_in[base + 6];  w.bias2 = (const float*)d_in[base + 7];
    w.g1   = (const float*)d_in[base + 8];  w.b1    = (const float*)d_in[base + 9];
    w.g2   = (const float*)d_in[base + 10]; w.b2    = (const float*)d_in[base + 11];
    return w;
  };
  LP sgw  = getLP(7);
  LP mg0w = getLP(19);
  LP mg1w = getLP(31);
  const float* pool_W = (const float*)d_in[43];
  const float* pool_b = (const float*)d_in[44];
  const float* pool_v = (const float*)d_in[45];
  const float* upd_W  = (const float*)d_in[46];
  const float* upd_b  = (const float*)d_in[47];
  const float* Wih    = (const float*)d_in[48];
  const float* Whh    = (const float*)d_in[49];
  const float* bih    = (const float*)d_in[50];
  const float* bhh    = (const float*)d_in[51];

  // ---- workspace carve (floats) ----
  const size_t MMAX = (size_t)B_ * 11 * 128;  // 5632 rows
  float* base = (float*)d_ws;
  size_t off = 0;
  auto alloc = [&](size_t n) { float* p = base + off; off += n; return p; };
  float* sel    = alloc((size_t)B_ * T_ * LL * HH);  // 786432
  float* ybuf   = alloc(MMAX * HH);
  float* qkvb   = alloc(MMAX * 3 * HH);
  float* tmpb   = alloc(MMAX * HH);
  float* projb  = alloc(MMAX * HH);
  float* xb     = alloc(MMAX * HH);
  float* h1b    = alloc(MMAX * FF);
  float* logits = alloc((size_t)B_ * LL * HH);
  float* ymean  = alloc((size_t)B_ * LL * HH);
  float* tanhb  = alloc((size_t)B_ * LL * HH);
  float* alphab = alloc(256);
  float* scoreb = alloc(64);
  float* pspk   = alloc(16);
  float* pcnt   = alloc(16);
  float* edub   = alloc((size_t)B_ * HH);
  float* holdb  = alloc((size_t)B_ * HH);
  float* gib    = alloc((size_t)B_ * 3 * HH);
  float* ghb    = alloc((size_t)B_ * 3 * HH);
  float* memb   = alloc((size_t)B_ * SM_ * HH);
  float* flopb  = alloc(16);

  unsigned short* bfp = (unsigned short*)(base + off);
  auto allocbf = [&](size_t n) { unsigned short* p = bfp; bfp += n; return p; };
  struct LB { unsigned short *qkv, *o, *w1, *w2; };
  auto allocLB = [&]() {
    LB b2;
    b2.qkv = allocbf((size_t)3 * HH * HH);
    b2.o   = allocbf((size_t)HH * HH);
    b2.w1  = allocbf((size_t)FF * HH);
    b2.w2  = allocbf((size_t)HH * FF);
    return b2;
  };
  LB sgb = allocLB(), mg0b = allocLB(), mg1b = allocLB();
  unsigned short* poolWbf = allocbf((size_t)HH * HH);
  unsigned short* Wihbf   = allocbf((size_t)3 * HH * HH);
  unsigned short* Whhbf   = allocbf((size_t)3 * HH * HH);

  auto cvt = [&](const float* s, unsigned short* d, int n) {
    k_cvt_bf16<<<(n + 255) / 256, 256, 0, stream>>>(s, d, n);
  };
  auto cvtL = [&](const LP& w, const LB& b2) {
    cvt(w.Wqkv, b2.qkv, 3 * HH * HH);
    cvt(w.Wo,   b2.o,   HH * HH);
    cvt(w.W1,   b2.w1,  FF * HH);
    cvt(w.W2,   b2.w2,  HH * FF);
  };
  cvtL(sgw, sgb); cvtL(mg0w, mg0b); cvtL(mg1w, mg1b);
  cvt(pool_W, poolWbf, HH * HH);
  cvt(Wih, Wihbf, 3 * HH * HH);
  cvt(Whh, Whhbf, 3 * HH * HH);

  auto gemm = [&](const float* A, const unsigned short* Wb, const float* bias, float* D,
                  int M, int N, int K, int act) {
    dim3 g((unsigned)((M + 15) / 16), (unsigned)(N / 256));
    k_gemm_bias<<<g, 128, 0, stream>>>(A, Wb, bias, D, M, N, K, act);
  };

  auto run_layer = [&](float* y, int nSeq, int S, const LP& w, const LB& wb) {
    int M = nSeq * S;
    gemm(y, wb.qkv, w.bqkv, qkvb, M, 3 * HH, HH, 0);
    if (S == 64)  k_attn<64><<<dim3(nSeq * NHH, 4), 32, 0, stream>>>(qkvb, tmpb);
    else          k_attn<128><<<dim3(nSeq * NHH, 8), 32, 0, stream>>>(qkvb, tmpb);
    gemm(tmpb, wb.o, w.bo, projb, M, HH, HH, 0);
    k_add_ln<<<M, 256, 0, stream>>>(y, projb, w.g1, w.b1, xb);
    gemm(xb, wb.w1, w.bias1, h1b, M, FF, HH, 1);      // fused exact-GELU
    gemm(h1b, wb.w2, w.bias2, projb, M, HH, FF, 0);
    k_add_ln<<<M, 256, 0, stream>>>(xb, projb, w.g2, w.b2, y);
  };

  // ---- init state ----
  k_zerof<<<8, 256, 0, stream>>>(memb, B_ * SM_ * HH);
  k_zerof<<<1, 32, 0, stream>>>(flopb, 1);
  k_embed<<<1024, 256, 0, stream>>>(ids, spk, tok_emb, tok_pe, spk_emb, sel);

  for (int l = 0; l < T_; ++l) {
    int lp = (l == 0) ? 1 : l;
    // speaker gate on cur = sel[:, l]
    k_slice_cur<<<256, 256, 0, stream>>>(sel, l, logits);
    run_layer(logits, B_, 64, sgw, sgb);
    k_spk_scores<<<B_ * SM_, 64, 0, stream>>>(memb, logits, scoreb);
    k_spk_soft<<<1, 32, 0, stream>>>(scoreb, spk, l, pspk, flopb);
    // build rel + pe
    {
      size_t tot = (size_t)B_ * lp * 128 * HH;
      int blocks = (int)((tot + 255) / 256);
      if (blocks > 2048) blocks = 2048;
      k_build_rel<<<blocks, 256, 0, stream>>>(sel, edu_pe, ybuf, l, lp);
    }
    // main gate x2
    run_layer(ybuf, B_ * lp, 128, mg0w, mg0b);
    run_layer(ybuf, B_ * lp, 128, mg1w, mg1b);
    // mean over lp -> output slice + ws
    k_ymean<<<256, 256, 0, stream>>>(ybuf, lp, l, (float*)d_out, ymean);
    // pooling
    gemm(ymean, poolWbf, pool_b, tanhb, B_ * LL, HH, HH, 2);   // fused tanh
    k_pool_score<<<B_, 64, 0, stream>>>(tanhb, pool_v, amask, l, alphab);
    k_pool_combine<<<B_, 256, 0, stream>>>(alphab, ymean, edub);
    k_updprob<<<B_, 256, 0, stream>>>(edub, upd_W, upd_b, pcnt);
    // GRU memory update
    k_gather_hold<<<B_, 256, 0, stream>>>(memb, spk, l, holdb);
    gemm(edub,  Wihbf, bih, gib, B_, 3 * HH, HH, 0);
    gemm(holdb, Whhbf, bhh, ghb, B_, 3 * HH, HH, 0);
    k_gru<<<B_, 256, 0, stream>>>(gib, ghb, holdb, pspk, pcnt, spk, l, memb);
  }
  k_finalize<<<1, 1, 0, stream>>>(flopb, (float*)d_out + (size_t)B_ * T_ * LL * HH);
}
